// MultiHeadAttention_86998857548086
// MI455X (gfx1250) — matrix operand order repaired and checked by reading, not compiled, not run on hardware
//
#include <hip/hip_runtime.h>
#include <hip/hip_bf16.h>

typedef __bf16 bf16_t;
typedef __attribute__((ext_vector_type(16))) __bf16 v16bf;
typedef __attribute__((ext_vector_type(8)))  __bf16 v8bf;
typedef __attribute__((ext_vector_type(8)))  float  v8f;
typedef int i32x4 __attribute__((vector_size(16)));

// ---------------------------------------------------------------------------
// Async global->LDS copy (GLOBAL_LOAD_ASYNC_TO_LDS_B128, ASYNCcnt) with
// synchronous fallback if the builtin is unavailable.
// ---------------------------------------------------------------------------
#if defined(__has_builtin)
#if __has_builtin(__builtin_amdgcn_global_load_async_to_lds_b128)
#define HAVE_ASYNC_LDS 1
#endif
#if __has_builtin(__builtin_amdgcn_s_wait_asynccnt)
#define HAVE_WAIT_ASYNC 1
#endif
#endif
#ifndef HAVE_ASYNC_LDS
#define HAVE_ASYNC_LDS 0
#endif
#ifndef HAVE_WAIT_ASYNC
#define HAVE_WAIT_ASYNC 0
#endif

__device__ inline void ld_async16(bf16_t* lds, const bf16_t* g) {
#if HAVE_ASYNC_LDS
  __builtin_amdgcn_global_load_async_to_lds_b128(
      (__attribute__((address_space(1))) i32x4*)g,
      (__attribute__((address_space(3))) i32x4*)lds, 0, 0);
#else
  *reinterpret_cast<uint4*>(lds) = *reinterpret_cast<const uint4*>(g);
#endif
}

__device__ inline void async_wait() {
#if HAVE_ASYNC_LDS
#if HAVE_WAIT_ASYNC
  __builtin_amdgcn_s_wait_asynccnt(0);
#else
  asm volatile("s_wait_asynccnt 0x0" ::: "memory");
#endif
#endif
}

// ---------------------------------------------------------------------------
// WMMA helpers (CDNA5 wave32, V_WMMA_F32_16X16X32_BF16)
// ---------------------------------------------------------------------------
__device__ inline v8f wmma_bf16(v16bf a, v16bf b, v8f c) {
  return __builtin_amdgcn_wmma_f32_16x16x32_bf16(false, a, false, b, (short)0, c,
                                                 false, false);
}

// A-matrix 16x32 bf16 fragment from row-major [M][K] LDS tile (ldm mult of 8).
__device__ inline v16bf load_frag_a(const bf16_t* tile, int ldm, int lane) {
  const int m = lane & 15;
  const int khalf = (lane & 16) >> 1;  // 0 or 8
  const bf16_t* p = tile + m * ldm + khalf;
  const v8bf lo = *reinterpret_cast<const v8bf*>(p);
  const v8bf hi = *reinterpret_cast<const v8bf*>(p + 16);
  return __builtin_shufflevector(lo, hi, 0, 1, 2, 3, 4, 5, 6, 7,
                                 8, 9, 10, 11, 12, 13, 14, 15);
}

// B-matrix 32x16 bf16 fragment from COLUMN-major [N][K] LDS tile (ldk mult 8).
__device__ inline v16bf load_frag_bt(const bf16_t* tile, int ldk, int lane) {
  const int n = lane & 15;
  const int kb = (lane & 16) >> 1;  // 0 or 8
  const bf16_t* p = tile + n * ldk + kb;
  const v8bf lo = *reinterpret_cast<const v8bf*>(p);
  const v8bf hi = *reinterpret_cast<const v8bf*>(p + 16);
  return __builtin_shufflevector(lo, hi, 0, 1, 2, 3, 4, 5, 6, 7,
                                 8, 9, 10, 11, 12, 13, 14, 15);
}

// fp32 -> bf16 4-element convert-store (for fp32 activation staging only)
__device__ inline void cvt4(bf16_t* dst, const float* src) {
  const float4 v = *reinterpret_cast<const float4*>(src);
  dst[0] = (bf16_t)v.x;
  dst[1] = (bf16_t)v.y;
  dst[2] = (bf16_t)v.z;
  dst[3] = (bf16_t)v.w;
}

// ---------------------------------------------------------------------------
// One-shot weight transpose: W[512][512] fp32 -> Wt[512][512] bf16 ([N][K]).
// ---------------------------------------------------------------------------
__global__ __launch_bounds__(256) void transpose_w4_kernel(
    const float* __restrict__ w0, const float* __restrict__ w1,
    const float* __restrict__ w2, const float* __restrict__ w3,
    bf16_t* __restrict__ o0, bf16_t* __restrict__ o1,
    bf16_t* __restrict__ o2, bf16_t* __restrict__ o3) {
  __shared__ float tile[32][33];
  const float* W;
  bf16_t* O;
  switch (blockIdx.z) {
    case 0: W = w0; O = o0; break;
    case 1: W = w1; O = o1; break;
    case 2: W = w2; O = o2; break;
    default: W = w3; O = o3; break;
  }
  const int n0 = blockIdx.x * 32;
  const int k0 = blockIdx.y * 32;
  const int tx = threadIdx.x, ty = threadIdx.y;
#pragma unroll
  for (int i = 0; i < 32; i += 8)
    tile[ty + i][tx] = W[(size_t)(k0 + ty + i) * 512 + n0 + tx];
  __syncthreads();
#pragma unroll
  for (int i = 0; i < 32; i += 8)
    O[(size_t)(n0 + ty + i) * 512 + k0 + tx] = (bf16_t)tile[tx][ty + i];
}

// ---------------------------------------------------------------------------
// Tiled GEMM with bias:  C[M,N] = A[M,K] @ W[K,N] + bias[N]
// Block 256 thr = 8 waves (4 row x 2 col groups), tile 128x128, BK 32.
// Wt pre-transposed bf16 [N][K] staged via async-to-LDS; fp32 A via VALU cvt,
// bf16 A via async-to-LDS. Each wave: 2x4 fragments = 8 WMMA per k-step.
// VT=true: scatter output in V-transposed layout [B][H][64][4096].
// ---------------------------------------------------------------------------
constexpr int BM = 128, BN = 128, BK = 32;
constexpr int LDK = BK + 8;  // 40 elems (80B rows, 16B aligned)

// fp32 source: 1024 4-elem chunks, 4 per thread, convert in VALU
__device__ inline void stage_a_tile(bf16_t* dstLds, const float* src, int t,
                                    int rowstride) {
#pragma unroll
  for (int i = 0; i < 4; ++i) {
    const int ch = t + i * 256;
    const int r = ch >> 3;
    const int c = (ch & 7) << 2;
    cvt4(&dstLds[r * LDK + c], src + (size_t)r * rowstride + c);
  }
}
// bf16 source: 512 16-byte chunks, 2 per thread, async direct to LDS
__device__ inline void stage_a_tile(bf16_t* dstLds, const bf16_t* src, int t,
                                    int rowstride) {
#pragma unroll
  for (int i = 0; i < 2; ++i) {
    const int ch = t + i * 256;
    const int r = ch >> 2;
    const int c = (ch & 3) << 3;
    ld_async16(&dstLds[r * LDK + c], src + (size_t)r * rowstride + c);
  }
}

template <typename AT, typename CT, bool VT>
__global__ __launch_bounds__(256) void gemm_bias_kernel(
    const AT* __restrict__ A, const bf16_t* __restrict__ Wt,
    const float* __restrict__ bias, CT* __restrict__ C, int M, int N, int K) {
  __shared__ alignas(16) bf16_t As[BM * LDK];
  __shared__ alignas(16) bf16_t Bs[BN * LDK];  // [n][k]
  const int t = threadIdx.x;
  const int lane = t & 31;
  const int wave = t >> 5;   // 0..7
  const int wr = wave >> 1;  // 0..3
  const int wc = wave & 1;   // 0..1
  const int row0 = blockIdx.y * BM;
  const int col0 = blockIdx.x * BN;

  v8f acc[2][4] = {};

  for (int k0 = 0; k0 < K; k0 += BK) {
    stage_a_tile(As, A + (size_t)row0 * K + k0, t, K);
    stage_a_tile(Bs, Wt + (size_t)col0 * K + k0, t, K);  // bf16 overload: async
    async_wait();
    __syncthreads();

    const v16bf a0 = load_frag_a(As + (wr * 32 + 0) * LDK, LDK, lane);
    const v16bf a1 = load_frag_a(As + (wr * 32 + 16) * LDK, LDK, lane);
#pragma unroll
    for (int f = 0; f < 4; ++f) {
      const v16bf b = load_frag_bt(Bs + (wc * 64 + f * 16) * LDK, LDK, lane);
      acc[0][f] = wmma_bf16(a0, b, acc[0][f]);
      acc[1][f] = wmma_bf16(a1, b, acc[1][f]);
    }
    __syncthreads();
  }

  const int rbase = row0 + wr * 32 + ((lane & 16) >> 1);
  const int cl = lane & 15;
#pragma unroll
  for (int mi = 0; mi < 2; ++mi) {
#pragma unroll
    for (int f = 0; f < 4; ++f) {
      const int cc = col0 + wc * 64 + f * 16 + cl;
      const float bv = bias[cc];
#pragma unroll
      for (int j = 0; j < 8; ++j) {
        const int rr = rbase + mi * 16 + j;
        const float val = acc[mi][f][j] + bv;
        if (VT) {
          // [B][H][d][S]: rr = b*4096+s ; cc = h*64+d
          const int bb = rr >> 12;
          const int s = rr & 4095;
          const int h = cc >> 6;
          const int d = cc & 63;
          C[((((size_t)bb << 3) + h) * 64 + d) * 4096 + s] = (CT)val;
        } else {
          C[(size_t)rr * N + cc] = (CT)val;
        }
      }
    }
  }
}

// ---------------------------------------------------------------------------
// Flash attention: one block = 128 queries of one (batch, head).
// 256 threads = 8 waves; wave w owns query rows [16w, 16w+16).
// Q,K: bf16 [B*S, 512] (col = h*64+d).  Vt: bf16 [B][H][64][S].
// K/V/Q tiles staged with async-to-LDS (all bf16, contiguous 16B chunks).
// ---------------------------------------------------------------------------
constexpr int ATT_LD = 72;  // 64 + 8 pad (144B rows, 16B aligned)
// softmax in exp2 domain: 1/sqrt(64) * log2(e)
#define SM_SCALE 0.1803368801111204f

__global__ __launch_bounds__(256) void attention_kernel(
    const bf16_t* __restrict__ Q, const bf16_t* __restrict__ Kp,
    const bf16_t* __restrict__ Vt, bf16_t* __restrict__ O, int S) {
  __shared__ alignas(16) bf16_t Qs[128 * ATT_LD];  // [q][d]
  __shared__ alignas(16) bf16_t Ks[64 * ATT_LD];   // [key][d]  (B for Q@K^T)
  __shared__ alignas(16) bf16_t Vs[64 * ATT_LD];   // [d][key]  (B for P@V)
  __shared__ alignas(16) bf16_t Ps[128 * ATT_LD];  // [q][key]  (A for P@V)

  const int t = threadIdx.x;
  const int lane = t & 31;
  const int wave = t >> 5;  // 0..7
  const int q0 = blockIdx.x * 128;
  const int h = blockIdx.y;
  const int b = blockIdx.z;
  const size_t ldr = 512;

  // Stage Q tile 128x64 (1024 x 16B chunks, 4/thread) via async copy
  const bf16_t* qbase = Q + ((size_t)b * S + q0) * ldr + h * 64;
#pragma unroll
  for (int i = 0; i < 4; ++i) {
    const int ch = t + i * 256;
    const int r = ch >> 3;
    const int c = (ch & 7) << 3;
    ld_async16(&Qs[r * ATT_LD + c], qbase + (size_t)r * ldr + c);
  }
  async_wait();
  __syncthreads();

  const v16bf aq0 = load_frag_a(Qs + wave * 16 * ATT_LD + 0, ATT_LD, lane);
  const v16bf aq1 = load_frag_a(Qs + wave * 16 * ATT_LD + 32, ATT_LD, lane);

  float mrow[8], lrow[8];
  v8f accO[4] = {};
#pragma unroll
  for (int j = 0; j < 8; ++j) { mrow[j] = -1e30f; lrow[j] = 0.0f; }

  const bf16_t* vtbase = Vt + (((size_t)b << 3) + h) * 64 * (size_t)S;
  const int nkt = S >> 6;
  for (int kt = 0; kt < nkt; ++kt) {
    // Stage K tile [64 key][64 d] and V tile [64 d][64 key]: 512 chunks each
    const bf16_t* kb = Kp + ((size_t)b * S + kt * 64) * ldr + h * 64;
#pragma unroll
    for (int i = 0; i < 2; ++i) {
      const int ch = t + i * 256;
      const int r = ch >> 3;
      const int c = (ch & 7) << 3;
      ld_async16(&Ks[r * ATT_LD + c], kb + (size_t)r * ldr + c);
      ld_async16(&Vs[r * ATT_LD + c], vtbase + (size_t)r * S + kt * 64 + c);
    }
    async_wait();
    __syncthreads();

    // S tile = Q @ K^T  (K-dim = d_head = 64 -> two bf16 k-steps)
    v8f sfrag[4] = {};
#pragma unroll
    for (int f = 0; f < 4; ++f) {
      const v16bf b0 = load_frag_bt(Ks + f * 16 * ATT_LD + 0, ATT_LD, lane);
      sfrag[f] = wmma_bf16(aq0, b0, sfrag[f]);
      const v16bf b1 = load_frag_bt(Ks + f * 16 * ATT_LD + 32, ATT_LD, lane);
      sfrag[f] = wmma_bf16(aq1, b1, sfrag[f]);
    }

    // Online softmax (exp2 domain; row reductions in 16-lane halves)
#pragma unroll
    for (int j = 0; j < 8; ++j) {
      const float s0 = sfrag[0][j] * SM_SCALE;
      const float s1 = sfrag[1][j] * SM_SCALE;
      const float s2 = sfrag[2][j] * SM_SCALE;
      const float s3 = sfrag[3][j] * SM_SCALE;
      float mx = fmaxf(fmaxf(s0, s1), fmaxf(s2, s3));
      mx = fmaxf(mx, __shfl_xor(mx, 1, 32));
      mx = fmaxf(mx, __shfl_xor(mx, 2, 32));
      mx = fmaxf(mx, __shfl_xor(mx, 4, 32));
      mx = fmaxf(mx, __shfl_xor(mx, 8, 32));
      const float mnew = fmaxf(mrow[j], mx);
      const float alpha = exp2f(mrow[j] - mnew);
      mrow[j] = mnew;
      const float p0 = exp2f(s0 - mnew);
      const float p1 = exp2f(s1 - mnew);
      const float p2 = exp2f(s2 - mnew);
      const float p3 = exp2f(s3 - mnew);
      sfrag[0][j] = p0;
      sfrag[1][j] = p1;
      sfrag[2][j] = p2;
      sfrag[3][j] = p3;
      float rs = p0 + p1 + p2 + p3;
      rs += __shfl_xor(rs, 1, 32);
      rs += __shfl_xor(rs, 2, 32);
      rs += __shfl_xor(rs, 4, 32);
      rs += __shfl_xor(rs, 8, 32);
      lrow[j] = lrow[j] * alpha + rs;
      accO[0][j] *= alpha;
      accO[1][j] *= alpha;
      accO[2][j] *= alpha;
      accO[3][j] *= alpha;
    }

    // Stage P (C-layout -> LDS -> A-layout)
    {
      const int prow = wave * 16 + ((lane & 16) >> 1);
      const int cl = lane & 15;
#pragma unroll
      for (int f = 0; f < 4; ++f)
#pragma unroll
        for (int j = 0; j < 8; ++j)
          Ps[(prow + j) * ATT_LD + f * 16 + cl] = (bf16_t)sfrag[f][j];
    }
    __syncthreads();

    // O += P @ V  (K-dim = key tile 64 -> two k-steps; B from Vt [d][key])
#pragma unroll
    for (int kk = 0; kk < 2; ++kk) {
      const v16bf a = load_frag_a(Ps + wave * 16 * ATT_LD + kk * 32, ATT_LD, lane);
#pragma unroll
      for (int f = 0; f < 4; ++f) {
        const v16bf bb = load_frag_bt(Vs + f * 16 * ATT_LD + kk * 32, ATT_LD, lane);
        accO[f] = wmma_bf16(a, bb, accO[f]);
      }
    }
    __syncthreads();
  }

  // Normalize and write (layout [B*S, 512], col = h*64 + d)
  const int rb = wave * 16 + ((lane & 16) >> 1);
  const int cl = lane & 15;
  bf16_t* obase = O + ((size_t)b * S + q0) * ldr + h * 64;
#pragma unroll
  for (int j = 0; j < 8; ++j) {
    const float inv = 1.0f / lrow[j];
#pragma unroll
    for (int f = 0; f < 4; ++f)
      obase[(size_t)(rb + j) * ldr + f * 16 + cl] = (bf16_t)(accO[f][j] * inv);
  }
}

// ---------------------------------------------------------------------------
// Launch: weight transpose -> Q/K/V projections -> flash attention -> out proj
// ---------------------------------------------------------------------------
extern "C" void kernel_launch(void* const* d_in, const int* in_sizes, int n_in,
                              void* d_out, int out_size, void* d_ws, size_t ws_size,
                              hipStream_t stream) {
  (void)in_sizes; (void)n_in; (void)out_size; (void)ws_size;
  const float* v_q = (const float*)d_in[0];
  const float* v_k = (const float*)d_in[1];
  const float* v_v = (const float*)d_in[2];
  const float* w_q = (const float*)d_in[3];
  const float* b_q = (const float*)d_in[4];
  const float* w_k = (const float*)d_in[5];
  const float* b_k = (const float*)d_in[6];
  const float* w_v = (const float*)d_in[7];
  const float* b_v = (const float*)d_in[8];
  const float* w_o = (const float*)d_in[9];
  const float* b_o = (const float*)d_in[10];

  constexpr int B = 2, S = 4096, D = 512;
  constexpr int M = B * S;  // 8192

  bf16_t* qws = (bf16_t*)d_ws;
  bf16_t* kws = qws + (size_t)M * D;
  bf16_t* vtws = kws + (size_t)M * D;  // [B][H][64][S]
  bf16_t* aws = vtws + (size_t)M * D;
  bf16_t* wtq = aws + (size_t)M * D;
  bf16_t* wtk = wtq + (size_t)D * D;
  bf16_t* wtv = wtk + (size_t)D * D;
  bf16_t* wto = wtv + (size_t)D * D;

  transpose_w4_kernel<<<dim3(16, 16, 4), dim3(32, 8), 0, stream>>>(
      w_q, w_k, w_v, w_o, wtq, wtk, wtv, wto);

  const dim3 gg(D / BN, M / BM);  // 4 x 64
  gemm_bias_kernel<float, bf16_t, false>
      <<<gg, 256, 0, stream>>>(v_q, wtq, b_q, qws, M, D, D);
  gemm_bias_kernel<float, bf16_t, false>
      <<<gg, 256, 0, stream>>>(v_k, wtk, b_k, kws, M, D, D);
  gemm_bias_kernel<float, bf16_t, true>
      <<<gg, 256, 0, stream>>>(v_v, wtv, b_v, vtws, M, D, D);

  const dim3 ga(S / 128, 8 /*heads*/, B);
  attention_kernel<<<ga, 256, 0, stream>>>(qws, kws, vtws, aws, S);

  gemm_bias_kernel<bf16_t, float, false>
      <<<gg, 256, 0, stream>>>(aws, wto, b_o, (float*)d_out, M, D, D);
}
